// FinalModel_46162308498185
// MI455X (gfx1250) — matrix operand order repaired and checked by reading, not compiled, not run on hardware
//
#include <hip/hip_runtime.h>
#include <hip/hip_bf16.h>

typedef __bf16 bf16_t;
typedef __attribute__((ext_vector_type(16))) __bf16 v16bf;
typedef __attribute__((ext_vector_type(8)))  __bf16 v8bf;
typedef __attribute__((ext_vector_type(8)))  float  v8f;

#define NB   512
#define SEQ  180
#define TP   384      // padded tokens per batch: 2 sources x 192
#define NH   256
#define PI_F    3.14159265358979323846f
#define HALF_PI 1.57079632679489662f

// ---------------- WMMA fragment loaders (CDNA5 16x16x32 bf16 layouts) -------

// A (16xK tile, row-major, ld elements): lane L: m=L&15, h=L>>4
// elements e: K = h*16 + e (contiguous)
__device__ __forceinline__ v16bf load_a_frag(const bf16_t* tile, int ld) {
  int lane = threadIdx.x & 31;
  int m = lane & 15, h = lane >> 4;
  const bf16_t* p = tile + (size_t)m * ld + h * 16;
  v8bf lo = *(const v8bf*)(p);
  v8bf hi = *(const v8bf*)(p + 8);
  return __builtin_shufflevector(lo, hi, 0,1,2,3,4,5,6,7,8,9,10,11,12,13,14,15);
}

// B (Kx16 logical tile) stored transposed [N][K] row-major at (n0,k0):
// lane L: n=L&15, h=L>>4, elements e: K = h*16 + e  (contiguous 32B)
__device__ __forceinline__ v16bf load_b_frag(const bf16_t* nt_base, int ldk) {
  int lane = threadIdx.x & 31;
  int n = lane & 15, h = lane >> 4;
  return *(const v16bf*)(nt_base + (size_t)n * ldk + h * 16);
}

// C/D: lane L: n = L&15 ; element r: m = r + 8*(L>>4)

// 16 x NH output tile, K = NH, A row-major (lda), W stored [N][K] bf16.
template <typename Epi>
__device__ __forceinline__ void gemm16xNHxNH(const bf16_t* Atile, int lda,
                                             const bf16_t* WT, Epi epi) {
  v16bf a[8];
#pragma unroll
  for (int kk = 0; kk < 8; ++kk) a[kk] = load_a_frag(Atile + kk * 32, lda);
#pragma unroll 1
  for (int nt = 0; nt < 16; ++nt) {
    v8f acc = {};
#pragma unroll
    for (int kk = 0; kk < 8; ++kk) {
      v16bf bfr = load_b_frag(WT + (size_t)nt * 16 * NH + kk * 32, NH);
      acc = __builtin_amdgcn_wmma_f32_16x16x32_bf16(false, a[kk], false, bfr,
                                                    (short)0, acc, false, false);
    }
    epi(nt, acc);
  }
}

// ---------------- K0: weight transpose f32[K][N] -> bf16[N][K] --------------
__global__ void k_transpose_bf16(const float* __restrict__ W, bf16_t* __restrict__ WT) {
  int k = blockIdx.x;    // input row
  int n = threadIdx.x;   // input col
  WT[(size_t)n * NH + k] = (bf16_t)W[(size_t)k * NH + n];
}

// ---------------- K1: xc constant embedding (VALU; tiny) --------------------
__global__ void k_xc_embed(const float* __restrict__ xc, const float* __restrict__ Wc1,
                           const float* __restrict__ bc1, const float* __restrict__ Wc2,
                           const float* __restrict__ bc2, float* __restrict__ xce) {
  __shared__ float xcs[16];
  __shared__ float h1[NH];
  int b = blockIdx.x, n = threadIdx.x;
  if (n < 16) xcs[n] = xc[b * 16 + n];
  __syncthreads();
  float acc = bc1[n];
  for (int k = 0; k < 16; ++k) acc += xcs[k] * Wc1[k * NH + n];
  h1[n] = fmaxf(acc, 0.f);
  __syncthreads();
  float o = bc2[n];
  for (int k = 0; k < NH; ++k) o += h1[k] * Wc2[k * NH + n];
  xce[(size_t)b * NH + n] = o;
}

// ---------------- K2: encoder (feature MLP + W2 WMMA + epilogue) ------------
__global__ void k_encode(const float* __restrict__ x0, const float* __restrict__ x1,
                         const float* __restrict__ pos0, const float* __restrict__ pos1,
                         const float* __restrict__ src_emb, const float* __restrict__ xce,
                         const float* __restrict__ W1_0, const float* __restrict__ b1_0,
                         const float* __restrict__ b2_0,
                         const float* __restrict__ W1_1, const float* __restrict__ b1_1,
                         const float* __restrict__ b2_1,
                         const bf16_t* __restrict__ W2_0T, const bf16_t* __restrict__ W2_1T,
                         float* __restrict__ xinf, bf16_t* __restrict__ xinb) {
  __shared__ bf16_t h1s[16 * NH];
  __shared__ float  poss[16];
  int b = blockIdx.x / 24, t = blockIdx.x % 24;
  int src = (t >= 12) ? 1 : 0;
  int r0 = t * 16;
  int lane = threadIdx.x;
  const float* W1 = src ? W1_1 : W1_0;
  const float* b1 = src ? b1_1 : b1_0;
  const float* b2 = src ? b2_1 : b2_0;
  int NF = src ? 8 : 12;

  for (int idx = lane; idx < 16 * NH; idx += 32) {
    int m = idx >> 8, n = idx & 255;
    int s = r0 + m - src * 192;
    float acc = 0.f;
    if (s < SEQ) {
      acc = b1[n];
      const float* xr = src ? (x1 + ((size_t)b * SEQ + s) * 8)
                            : (x0 + ((size_t)b * SEQ + s) * 12);
      for (int k = 0; k < NF; ++k) acc += xr[k] * W1[k * NH + n];
      acc = fmaxf(acc, 0.f);
    }
    h1s[m * NH + n] = (bf16_t)acc;
  }
  if (lane < 16) {
    int s = r0 + lane - src * 192;
    poss[lane] = (s < SEQ) ? (src ? pos1[b * SEQ + s] : pos0[b * SEQ + s]) : 0.f;
  }
  __syncthreads();

  const bf16_t* WT = src ? W2_1T : W2_0T;
  gemm16xNHxNH(h1s, NH, WT, [&](int nt, v8f acc) {
    int n = nt * 16 + (lane & 15);
    float base = b2[n] + src_emb[src * NH + n] + xce[(size_t)b * NH + n];
    // pi/(n/2+1) via hardware v_rcp (plenty for a positional encoding);
    // cos(x) == sin(x + pi/2): single v_sin, no EXEC divergence.
    float inv   = PI_F * __builtin_amdgcn_rcpf((float)(n / 2 + 1));
    float phase = (n & 1) ? HALF_PI : 0.f;
#pragma unroll
    for (int rr = 0; rr < 8; ++rr) {
      int m = (lane >> 4) * 8 + rr;
      int r = r0 + m;
      int s = r - src * 192;
      float pe = __sinf(poss[m] * inv + phase);
      float val = (s < SEQ) ? (acc[rr] + base + pe) : 0.f;
      size_t off = ((size_t)b * TP + r) * NH + n;
      xinf[off] = val;
      xinb[off] = (bf16_t)val;
    }
  });
}

// ---------------- K3: Q/K/V projections (V stored transposed) ---------------
__global__ void k_qkv(const bf16_t* __restrict__ xinb,
                      const bf16_t* __restrict__ WqT, const bf16_t* __restrict__ WkT,
                      const bf16_t* __restrict__ WvT,
                      bf16_t* __restrict__ qb, bf16_t* __restrict__ kb,
                      bf16_t* __restrict__ vt) {
  int b = blockIdx.x / 24, t = blockIdx.x % 24;
  int r0 = t * 16;
  int lane = threadIdx.x;
  const bf16_t* A = xinb + ((size_t)b * TP + r0) * NH;
  gemm16xNHxNH(A, NH, WqT, [&](int nt, v8f acc) {
    int n = nt * 16 + (lane & 15);
#pragma unroll
    for (int rr = 0; rr < 8; ++rr)
      qb[((size_t)b * TP + r0 + (lane >> 4) * 8 + rr) * NH + n] = (bf16_t)acc[rr];
  });
  gemm16xNHxNH(A, NH, WkT, [&](int nt, v8f acc) {
    int n = nt * 16 + (lane & 15);
#pragma unroll
    for (int rr = 0; rr < 8; ++rr)
      kb[((size_t)b * TP + r0 + (lane >> 4) * 8 + rr) * NH + n] = (bf16_t)acc[rr];
  });
  gemm16xNHxNH(A, NH, WvT, [&](int nt, v8f acc) {
    int n = nt * 16 + (lane & 15);
    v8bf pv;
#pragma unroll
    for (int rr = 0; rr < 8; ++rr) pv[rr] = (bf16_t)acc[rr];
    *(v8bf*)(vt + ((size_t)b * NH + n) * TP + r0 + (lane >> 4) * 8) = pv;
  });
}

// ---------------- K4: fused attention (scores -> softmax -> P@V) ------------
__global__ void k_attn(const bf16_t* __restrict__ qb, const bf16_t* __restrict__ kb,
                       const bf16_t* __restrict__ vt, bf16_t* __restrict__ atto) {
  __shared__ float  sc[16 * TP];
  __shared__ bf16_t pb[16 * TP];
  int b = blockIdx.x / 24, qt = blockIdx.x % 24;
  int r0 = qt * 16;
  int lane = threadIdx.x;
  const float scale = 0.05270462766947299f;  // 1/sqrt(360) (seq-len quirk)

  v16bf a[8];
#pragma unroll
  for (int kk = 0; kk < 8; ++kk)
    a[kk] = load_a_frag(qb + ((size_t)b * TP + r0) * NH + kk * 32, NH);

  const bf16_t* kbase = kb + (size_t)b * TP * NH;
#pragma unroll 1
  for (int kt = 0; kt < 24; ++kt) {
    v8f acc = {};
#pragma unroll
    for (int kk = 0; kk < 8; ++kk) {
      v16bf bfr = load_b_frag(kbase + (size_t)(kt * 16) * NH + kk * 32, NH);
      acc = __builtin_amdgcn_wmma_f32_16x16x32_bf16(false, a[kk], false, bfr,
                                                    (short)0, acc, false, false);
    }
    int c = kt * 16 + (lane & 15);
    int cs = c - ((c >= 192) ? 192 : 0);
    bool cvalid = cs < SEQ;
#pragma unroll
    for (int rr = 0; rr < 8; ++rr) {
      int m = (lane >> 4) * 8 + rr;
      sc[m * TP + c] = cvalid ? acc[rr] * scale : -1e30f;
    }
  }
  __syncthreads();

  // softmax: 2 lanes per row, 192 columns each
  int row = lane & 15, half = lane >> 4;
  float* srow = sc + row * TP + half * 192;
  float mx = -1e30f;
  for (int c = 0; c < 192; ++c) mx = fmaxf(mx, srow[c]);
  mx = fmaxf(mx, __shfl_xor(mx, 16));
  float sum = 0.f;
  for (int c = 0; c < 192; ++c) { float e = __expf(srow[c] - mx); srow[c] = e; sum += e; }
  sum += __shfl_xor(sum, 16);
  float inv = 1.f / sum;
  bf16_t* prow = pb + row * TP + half * 192;
  for (int c = 0; c < 192; ++c) prow[c] = (bf16_t)(srow[c] * inv);
  __syncthreads();

  // out = P[16x384] @ V[384x256]   (V stored transposed [ch][tok])
  v16bf ap[12];
#pragma unroll
  for (int ks = 0; ks < 12; ++ks) ap[ks] = load_a_frag(pb + ks * 32, TP);
  const bf16_t* vbase = vt + (size_t)b * NH * TP;
#pragma unroll 1
  for (int nt = 0; nt < 16; ++nt) {
    v8f acc = {};
#pragma unroll
    for (int ks = 0; ks < 12; ++ks) {
      v16bf bv = load_b_frag(vbase + (size_t)(nt * 16) * TP + ks * 32, TP);
      acc = __builtin_amdgcn_wmma_f32_16x16x32_bf16(false, ap[ks], false, bv,
                                                    (short)0, acc, false, false);
    }
    int n = nt * 16 + (lane & 15);
#pragma unroll
    for (int rr = 0; rr < 8; ++rr)
      atto[((size_t)b * TP + r0 + (lane >> 4) * 8 + rr) * NH + n] = (bf16_t)acc[rr];
  }
}

// ---------------- K5/K7: GEMM (+opt bias) + residual(f32) + LayerNorm -------
__global__ void k_gemm_res_ln(const bf16_t* __restrict__ Ain, const bf16_t* __restrict__ WT,
                              const float* __restrict__ bias,   // may be nullptr
                              const float* __restrict__ resf,
                              const float* __restrict__ g, const float* __restrict__ bb,
                              bf16_t* __restrict__ outb) {
  __shared__ float hs[16 * NH];
  int b = blockIdx.x / 24, t = blockIdx.x % 24;
  int r0 = t * 16;
  int lane = threadIdx.x;
  const bf16_t* A = Ain + ((size_t)b * TP + r0) * NH;
  gemm16xNHxNH(A, NH, WT, [&](int nt, v8f acc) {
    int n = nt * 16 + (lane & 15);
    float bv = bias ? bias[n] : 0.f;
#pragma unroll
    for (int rr = 0; rr < 8; ++rr) {
      int m = (lane >> 4) * 8 + rr;
      size_t off = ((size_t)b * TP + r0 + m) * NH + n;
      hs[m * NH + n] = acc[rr] + bv + resf[off];
    }
  });
  __syncthreads();
  int row = lane & 15, half = lane >> 4;
  const float* hr = hs + row * NH + half * 128;
  float s = 0.f, s2 = 0.f;
  for (int c = 0; c < 128; ++c) { float v = hr[c]; s += v; s2 += v * v; }
  s  += __shfl_xor(s, 16);
  s2 += __shfl_xor(s2, 16);
  float mean = s * (1.f / NH);
  float var  = s2 * (1.f / NH) - mean * mean;
  float rstd = rsqrtf(var + 1e-5f);
  bf16_t* orow = outb + ((size_t)b * TP + r0 + row) * NH + half * 128;
  const float* gg = g + half * 128; const float* bbb = bb + half * 128;
  for (int c = 0; c < 128; ++c)
    orow[c] = (bf16_t)((hr[c] - mean) * rstd * gg[c] + bbb[c]);
}

// ---------------- K6: GEMM + bias + relu ------------------------------------
__global__ void k_gemm_bias_relu(const bf16_t* __restrict__ Ain, const bf16_t* __restrict__ WT,
                                 const float* __restrict__ bias, bf16_t* __restrict__ outb) {
  int b = blockIdx.x / 24, t = blockIdx.x % 24;
  int r0 = t * 16;
  int lane = threadIdx.x;
  const bf16_t* A = Ain + ((size_t)b * TP + r0) * NH;
  gemm16xNHxNH(A, NH, WT, [&](int nt, v8f acc) {
    int n = nt * 16 + (lane & 15);
    float bv = bias[n];
#pragma unroll
    for (int rr = 0; rr < 8; ++rr)
      outb[((size_t)b * TP + r0 + (lane >> 4) * 8 + rr) * NH + n] =
          (bf16_t)fmaxf(acc[rr] + bv, 0.f);
  });
}

// ---------------- K8: final MLP head -> per-token scalar --------------------
__global__ void k_final(const bf16_t* __restrict__ h2, const bf16_t* __restrict__ f2W1T,
                        const float* __restrict__ f2b1, const float* __restrict__ f2W2,
                        const float* __restrict__ f2b2, float* __restrict__ y) {
  __shared__ float hs[16 * NH];
  int b = blockIdx.x / 24, t = blockIdx.x % 24;
  int r0 = t * 16;
  int lane = threadIdx.x;
  const bf16_t* A = h2 + ((size_t)b * TP + r0) * NH;
  gemm16xNHxNH(A, NH, f2W1T, [&](int nt, v8f acc) {
    int n = nt * 16 + (lane & 15);
    float bv = f2b1[n];
#pragma unroll
    for (int rr = 0; rr < 8; ++rr) {
      int m = (lane >> 4) * 8 + rr;
      hs[m * NH + n] = fmaxf(acc[rr] + bv, 0.f);
    }
  });
  __syncthreads();
  int row = lane & 15, half = lane >> 4;
  const float* hr = hs + row * NH + half * 128;
  const float* w  = f2W2 + half * 128;
  float s = 0.f;
  for (int c = 0; c < 128; ++c) s += hr[c] * w[c];
  s += __shfl_xor(s, 16);
  if (half == 0) y[(size_t)b * TP + r0 + row] = s + f2b2[0];
}

// ---------------- K9: segment mean pooling ----------------------------------
__global__ void k_pool(const float* __restrict__ y, float* __restrict__ out) {
  __shared__ float red[256];
  int b = blockIdx.x, tid = threadIdx.x;
  float v = 0.f;
  if (tid < SEQ) v = y[(size_t)b * TP + tid] + y[(size_t)b * TP + 192 + tid];
  red[tid] = v;
  __syncthreads();
  for (int s = 128; s > 0; s >>= 1) {
    if (tid < s) red[tid] += red[tid + s];
    __syncthreads();
  }
  if (tid == 0) out[b] = red[0] * (1.f / SEQ);
}

// ---------------- host launch ------------------------------------------------
extern "C" void kernel_launch(void* const* d_in, const int* in_sizes, int n_in,
                              void* d_out, int out_size, void* d_ws, size_t ws_size,
                              hipStream_t stream) {
  (void)in_sizes; (void)n_in; (void)out_size; (void)ws_size;
  const float* x0    = (const float*)d_in[0];
  const float* x1    = (const float*)d_in[1];
  const float* pos0  = (const float*)d_in[2];
  const float* pos1  = (const float*)d_in[3];
  const float* xc    = (const float*)d_in[4];
  const float* semb  = (const float*)d_in[5];
  const float* W1_0  = (const float*)d_in[6];
  const float* b1_0  = (const float*)d_in[7];
  const float* W2_0  = (const float*)d_in[8];
  const float* b2_0  = (const float*)d_in[9];
  const float* W1_1  = (const float*)d_in[10];
  const float* b1_1  = (const float*)d_in[11];
  const float* W2_1  = (const float*)d_in[12];
  const float* b2_1  = (const float*)d_in[13];
  const float* Wc1   = (const float*)d_in[14];
  const float* bc1   = (const float*)d_in[15];
  const float* Wc2   = (const float*)d_in[16];
  const float* bc2   = (const float*)d_in[17];
  const float* Wq    = (const float*)d_in[18];
  const float* Wk    = (const float*)d_in[19];
  const float* Wv    = (const float*)d_in[20];
  const float* Wo    = (const float*)d_in[21];
  const float* ln1_g = (const float*)d_in[22];
  const float* ln1_b = (const float*)d_in[23];
  const float* ln2_g = (const float*)d_in[24];
  const float* ln2_b = (const float*)d_in[25];
  const float* f1W1  = (const float*)d_in[26];
  const float* f1b1  = (const float*)d_in[27];
  const float* f1W2  = (const float*)d_in[28];
  const float* f1b2  = (const float*)d_in[29];
  const float* f2W1  = (const float*)d_in[30];
  const float* f2b1  = (const float*)d_in[31];
  const float* f2W2  = (const float*)d_in[32];
  const float* f2b2  = (const float*)d_in[33];

  // workspace layout (~579 MiB total)
  constexpr size_t WT_ELEMS = (size_t)NH * NH;
  constexpr size_t TOKS     = (size_t)NB * TP * NH;
  char* ws = (char*)d_ws;
  size_t off = 0;
  bf16_t* wt   = (bf16_t*)(ws + off); off += 9 * WT_ELEMS * sizeof(bf16_t);
  float*  xce  = (float*) (ws + off); off += (size_t)NB * NH * sizeof(float);
  float*  xinf = (float*) (ws + off); off += TOKS * sizeof(float);
  bf16_t* buf0 = (bf16_t*)(ws + off); off += TOKS * sizeof(bf16_t); // xIn_bf -> atto
  bf16_t* buf1 = (bf16_t*)(ws + off); off += TOKS * sizeof(bf16_t); // q      -> h
  bf16_t* buf2 = (bf16_t*)(ws + off); off += TOKS * sizeof(bf16_t); // k      -> ffn-mid
  bf16_t* buf3 = (bf16_t*)(ws + off); off += TOKS * sizeof(bf16_t); // vT     -> h2
  float*  ybuf = (float*) (ws + off); off += (size_t)NB * TP * sizeof(float);

  bf16_t* W2_0T = wt + 0 * WT_ELEMS;
  bf16_t* W2_1T = wt + 1 * WT_ELEMS;
  bf16_t* WqT   = wt + 2 * WT_ELEMS;
  bf16_t* WkT   = wt + 3 * WT_ELEMS;
  bf16_t* WvT   = wt + 4 * WT_ELEMS;
  bf16_t* WoT   = wt + 5 * WT_ELEMS;
  bf16_t* f1W1T = wt + 6 * WT_ELEMS;
  bf16_t* f1W2T = wt + 7 * WT_ELEMS;
  bf16_t* f2W1T = wt + 8 * WT_ELEMS;

  k_transpose_bf16<<<NH, NH, 0, stream>>>(W2_0, W2_0T);
  k_transpose_bf16<<<NH, NH, 0, stream>>>(W2_1, W2_1T);
  k_transpose_bf16<<<NH, NH, 0, stream>>>(Wq,   WqT);
  k_transpose_bf16<<<NH, NH, 0, stream>>>(Wk,   WkT);
  k_transpose_bf16<<<NH, NH, 0, stream>>>(Wv,   WvT);
  k_transpose_bf16<<<NH, NH, 0, stream>>>(Wo,   WoT);
  k_transpose_bf16<<<NH, NH, 0, stream>>>(f1W1, f1W1T);
  k_transpose_bf16<<<NH, NH, 0, stream>>>(f1W2, f1W2T);
  k_transpose_bf16<<<NH, NH, 0, stream>>>(f2W1, f2W1T);

  k_xc_embed<<<NB, NH, 0, stream>>>(xc, Wc1, bc1, Wc2, bc2, xce);

  dim3 grid(NB * 24);
  k_encode<<<grid, 32, 0, stream>>>(x0, x1, pos0, pos1, semb, xce,
                                    W1_0, b1_0, b2_0, W1_1, b1_1, b2_1,
                                    W2_0T, W2_1T, xinf, buf0);
  k_qkv<<<grid, 32, 0, stream>>>(buf0, WqT, WkT, WvT, buf1, buf2, buf3);
  k_attn<<<grid, 32, 0, stream>>>(buf1, buf2, buf3, buf0);
  k_gemm_res_ln<<<grid, 32, 0, stream>>>(buf0, WoT, nullptr, xinf, ln1_g, ln1_b, buf1);
  k_gemm_bias_relu<<<grid, 32, 0, stream>>>(buf1, f1W1T, f1b1, buf2);
  k_gemm_res_ln<<<grid, 32, 0, stream>>>(buf2, f1W2T, f1b2, xinf, ln2_g, ln2_b, buf3);
  k_final<<<grid, 32, 0, stream>>>(buf3, f2W1T, f2b1, f2W2, f2b2, ybuf);
  k_pool<<<NB, 256, 0, stream>>>(ybuf, (float*)d_out);
}
